// SS2D_MB_23192823398799
// MI455X (gfx1250) — matrix operand order repaired and checked by reading, not compiled, not run on hardware
//
#include <hip/hip_runtime.h>
#include <hip/hip_bf16.h>

typedef __attribute__((ext_vector_type(16))) _Float16 v16h;
typedef __attribute__((ext_vector_type(8)))  _Float16 v8h;
typedef __attribute__((ext_vector_type(8)))  float    v8f;

// ---------------- problem constants ----------------
constexpr int kB = 2;
constexpr int kH = 48;
constexpr int kW = 48;
constexpr int kD = 192;
constexpr int kC = 384;           // D * E
constexpr int kL = kH * kW;       // 2304
constexpr int kN = 16;
constexpr int kR = 12;
constexpr int kK = 4;             // scan directions
constexpr int kP = kB * kH * kW;  // 4608 pixels
constexpr int kG = 24;            // scan chunks
constexpr int kT = 96;            // steps per chunk (kG*kT == kL)

#define TILE_M 128
#define TILE_N 64
#define TILE_K 32

// ---------------- staging helpers (no guards: caller guarantees alignment) ------
__device__ __forceinline__ void ss2d_stage_a(const float* __restrict__ A, int lda,
                                             int m0, int k0, int ar, int ac,
                                             _Float16 (*As)[40])
{
    const float4* ap =
        reinterpret_cast<const float4*>(A + (long)(m0 + ar) * lda + (k0 + ac));
    float4 q0 = ap[0], q1 = ap[1], q2 = ap[2], q3 = ap[3];
    v8h p0, p1;
    p0[0] = (_Float16)q0.x; p0[1] = (_Float16)q0.y;
    p0[2] = (_Float16)q0.z; p0[3] = (_Float16)q0.w;
    p0[4] = (_Float16)q1.x; p0[5] = (_Float16)q1.y;
    p0[6] = (_Float16)q1.z; p0[7] = (_Float16)q1.w;
    p1[0] = (_Float16)q2.x; p1[1] = (_Float16)q2.y;
    p1[2] = (_Float16)q2.z; p1[3] = (_Float16)q2.w;
    p1[4] = (_Float16)q3.x; p1[5] = (_Float16)q3.y;
    p1[6] = (_Float16)q3.z; p1[7] = (_Float16)q3.w;
    *reinterpret_cast<v8h*>(&As[ar][ac])     = p0;
    *reinterpret_cast<v8h*>(&As[ar][ac + 8]) = p1;
}

template<int BT>
__device__ __forceinline__ void ss2d_stage_b(const float* __restrict__ Bsrc, int ldb,
                                             int n0, int k0, int bn, int bc,
                                             int uc, int un, _Float16 (*Bs)[40])
{
    if (BT) {  // Bsrc is [N][K] (weight-style): contiguous along K
        const float4* bp =
            reinterpret_cast<const float4*>(Bsrc + (long)(n0 + bn) * ldb + (k0 + bc));
        float4 q0 = bp[0], q1 = bp[1];
        v8h p0;
        p0[0] = (_Float16)q0.x; p0[1] = (_Float16)q0.y;
        p0[2] = (_Float16)q0.z; p0[3] = (_Float16)q0.w;
        p0[4] = (_Float16)q1.x; p0[5] = (_Float16)q1.y;
        p0[6] = (_Float16)q1.z; p0[7] = (_Float16)q1.w;
        *reinterpret_cast<v8h*>(&Bs[bn][bc]) = p0;
    } else {   // Bsrc is [K][N]: contiguous along N, scatter into n-major LDS
        const float4* bp =
            reinterpret_cast<const float4*>(Bsrc + (long)(k0 + uc) * ldb + (n0 + un));
        float4 q0 = bp[0], q1 = bp[1];
        Bs[un + 0][uc] = (_Float16)q0.x;
        Bs[un + 1][uc] = (_Float16)q0.y;
        Bs[un + 2][uc] = (_Float16)q0.z;
        Bs[un + 3][uc] = (_Float16)q0.w;
        Bs[un + 4][uc] = (_Float16)q1.x;
        Bs[un + 5][uc] = (_Float16)q1.y;
        Bs[un + 6][uc] = (_Float16)q1.z;
        Bs[un + 7][uc] = (_Float16)q1.w;
    }
}

// ---------------- generic WMMA GEMM (templated, double-buffered LDS) ----------
// C[M][N] = A[M][K] * B (+ resid).  A f32 row-major (lda=K).
// Requirements: A has >= ceil(M/128)*128 readable rows, N % 64 == 0, Kd % 32 == 0.
// Batched over blockIdx.z; A batch index = z % aMod.
template<int BT, int RESID>
__global__ __launch_bounds__(256)
void ss2d_gemm_wmma(const float* __restrict__ A, long aStride, int aMod,
                    const float* __restrict__ Bsrc, long bStride,
                    float* __restrict__ Cmat, long cStride,
                    const float* __restrict__ resid,
                    int M, int N, int Kd, int lda, int ldb, int ldc)
{
    __shared__ _Float16 As[2][TILE_M][40];
    __shared__ _Float16 Bs[2][TILE_N][40];

    const int z = blockIdx.z;
    A    += (long)(z % aMod) * aStride;
    Bsrc += (long)z * bStride;
    Cmat += (long)z * cStride;

    const int m0 = blockIdx.y * TILE_M;
    const int n0 = blockIdx.x * TILE_N;
    const int tid  = threadIdx.x;
    const int lane = tid & 31;
    const int wid  = tid >> 5;
    const int wm   = (wid & 3) * 32;   // wave row offset
    const int wn   = (wid >> 2) * 32;  // wave col offset
    const int lo   = lane & 15;
    const int hi   = lane >> 4;

    // per-thread staging coordinates (uniform across the loop)
    const int ar = tid >> 1;               // A row 0..127
    const int ac = (tid & 1) * 16;         // A col base 0 / 16
    const int bn = tid >> 2;               // B(t) row 0..63
    const int bc = (tid & 3) * 8;          // B(t) col base
    const int uc = tid >> 3;               // B(nt) k-row 0..31
    const int un = (tid & 7) * 8;          // B(nt) n base

    v8f acc[2][2];
    #pragma unroll
    for (int i = 0; i < 2; ++i)
        #pragma unroll
        for (int j = 0; j < 2; ++j)
            acc[i][j] = (v8f){0.f,0.f,0.f,0.f,0.f,0.f,0.f,0.f};

    // prologue: stage tile 0 into buffer 0
    ss2d_stage_a(A, lda, m0, 0, ar, ac, As[0]);
    ss2d_stage_b<BT>(Bsrc, ldb, n0, 0, bn, bc, uc, un, Bs[0]);
    __syncthreads();

    int cur = 0;
    for (int k0 = 0; k0 < Kd; k0 += TILE_K) {
        // stage next tile into the alternate buffer while this one computes
        if (k0 + TILE_K < Kd) {
            ss2d_stage_a(A, lda, m0, k0 + TILE_K, ar, ac, As[cur ^ 1]);
            ss2d_stage_b<BT>(Bsrc, ldb, n0, k0 + TILE_K, bn, bc, uc, un, Bs[cur ^ 1]);
            // speculative prefetch two tiles ahead (global_prefetch_b8)
            if (k0 + 2 * TILE_K < Kd)
                __builtin_prefetch(A + (long)(m0 + ar) * lda + (k0 + 2 * TILE_K + ac), 0, 0);
        }

        // ---- fragments: ISA 16-bit A layout / n-major B layout ----
        v16h afrag[2], bfrag[2];
        #pragma unroll
        for (int i = 0; i < 2; ++i) {
            int row = wm + i * 16 + lo;
            const _Float16* p0 = &As[cur][row][hi * 8];
            const _Float16* p1 = &As[cur][row][16 + hi * 8];
            #pragma unroll
            for (int j = 0; j < 8; ++j) {
                afrag[i][j]     = p0[j];
                afrag[i][8 + j] = p1[j];
            }
        }
        #pragma unroll
        for (int jn = 0; jn < 2; ++jn) {
            int col = wn + jn * 16 + lo;
            const _Float16* p0 = &Bs[cur][col][hi * 8];
            const _Float16* p1 = &Bs[cur][col][16 + hi * 8];
            #pragma unroll
            for (int j = 0; j < 8; ++j) {
                bfrag[jn][j]     = p0[j];
                bfrag[jn][8 + j] = p1[j];
            }
        }
        #pragma unroll
        for (int i = 0; i < 2; ++i)
            #pragma unroll
            for (int jn = 0; jn < 2; ++jn)
                acc[i][jn] = __builtin_amdgcn_wmma_f32_16x16x32_f16(
                    false, afrag[i], false, bfrag[jn],
                    (short)0, acc[i][jn], false, false);

        __syncthreads();   // single barrier per iteration (double-buffered)
        cur ^= 1;
    }

    // ---- epilogue: C/D layout (row = hi*8 + r, col = lo) ----
    const float* rptr = RESID ? (resid + (long)z * cStride) : nullptr;
    #pragma unroll
    for (int i = 0; i < 2; ++i) {
        #pragma unroll
        for (int jn = 0; jn < 2; ++jn) {
            int col = n0 + wn + jn * 16 + lo;
            #pragma unroll
            for (int r = 0; r < 8; ++r) {
                int row = m0 + wm + i * 16 + hi * 8 + r;
                if (row < M) {
                    long idx = (long)row * ldc + col;
                    float v = acc[i][jn][r];
                    if (RESID) v += rptr[idx];
                    Cmat[idx] = v;
                }
            }
        }
    }
}

// ---------------- pad x_proj_w (4 x 44 x 384) -> (4 x 128 x 384, zero-filled) ----
__global__ void ss2d_pad_xprojw(const float* __restrict__ src, float* __restrict__ dst)
{
    int k = blockIdx.y;
    int i = blockIdx.x * blockDim.x + threadIdx.x;   // over 128*384
    if (i < TILE_M * kC) {
        int r = i / kC, c = i % kC;
        float v = (r < 44) ? src[((long)k * 44 + r) * kC + c] : 0.0f;
        dst[(long)k * TILE_M * kC + i] = v;
    }
}

// ---------------- block reduction helper ----------------
__device__ __forceinline__ float ss2d_block_reduce(float* red, int tid, int n) {
    __syncthreads();
    if (n == 384) {
        if (tid < 128) red[tid] += red[tid + 128] + red[tid + 256];
        __syncthreads();
        if (tid < 64) red[tid] += red[tid + 64];
        __syncthreads();
    } else { // n == 192
        if (tid < 64) red[tid] += red[tid + 64] + red[tid + 128];
        __syncthreads();
    }
    if (tid < 32) red[tid] += red[tid + 32]; __syncthreads();
    if (tid < 16) red[tid] += red[tid + 16]; __syncthreads();
    if (tid <  8) red[tid] += red[tid +  8]; __syncthreads();
    if (tid <  4) red[tid] += red[tid +  4]; __syncthreads();
    if (tid <  2) red[tid] += red[tid +  2]; __syncthreads();
    if (tid <  1) red[tid] += red[tid +  1]; __syncthreads();
    return red[0];
}

// ---------------- LN (input) ----------------
__global__ __launch_bounds__(192)
void ss2d_ln_in(const float* __restrict__ x, const float* __restrict__ g,
                const float* __restrict__ b, float* __restrict__ h)
{
    __shared__ float red[192];
    __shared__ float stat;
    int p = blockIdx.x, d = threadIdx.x;
    float v = x[(long)p * kD + d];
    red[d] = v;
    float s = ss2d_block_reduce(red, d, 192);
    if (d == 0) stat = s / (float)kD;
    __syncthreads();
    float mean = stat;
    float e = v - mean;
    __syncthreads();
    red[d] = e * e;
    s = ss2d_block_reduce(red, d, 192);
    if (d == 0) stat = s / (float)kD;
    __syncthreads();
    float var = stat;
    h[(long)p * kD + d] = e * rsqrtf(var + 1e-5f) * g[d] + b[d];
}

// ---------------- depthwise conv 3x3 + bias + SiLU ----------------
__global__ __launch_bounds__(384)
void ss2d_conv_silu(const float* __restrict__ xz, const float* __restrict__ cw,
                    const float* __restrict__ cb, float* __restrict__ xconv)
{
    int pix = blockIdx.x;              // b*H*W + y*W + x
    int c = threadIdx.x;               // 0..383
    int b = pix / (kH * kW);
    int rem = pix % (kH * kW);
    int y = rem / kW, x = rem % kW;
    float acc = cb[c];
    #pragma unroll
    for (int dy = -1; dy <= 1; ++dy) {
        #pragma unroll
        for (int dx = -1; dx <= 1; ++dx) {
            int yy = y + dy, xx = x + dx;
            if (yy >= 0 && yy < kH && xx >= 0 && xx < kW) {
                float xv = xz[((long)b * kH * kW + yy * kW + xx) * (2 * kC) + c];
                float wv = cw[c * 9 + (dy + 1) * 3 + (dx + 1)];
                acc += xv * wv;
            }
        }
    }
    float s = acc / (1.f + __expf(-acc));
    xconv[((long)(b * kC + c)) * kL + rem] = s;
}

// ---------------- build 4 scan directions ----------------
__global__ void ss2d_build_xs(const float* __restrict__ xconv, float* __restrict__ xs)
{
    int bc = blockIdx.x;               // b*C + c
    int b = bc / kC, c = bc % kC;
    for (int l = threadIdx.x; l < kL; l += blockDim.x) {
        float v = xconv[(long)bc * kL + l];
        int y = l / kW, x = l % kW;
        int l1 = x * kH + y;
        xs[((long)(b * kK + 0) * kC + c) * kL + l]             = v;
        xs[((long)(b * kK + 1) * kC + c) * kL + l1]            = v;
        xs[((long)(b * kK + 2) * kC + c) * kL + (kL - 1 - l)]  = v;
        xs[((long)(b * kK + 3) * kC + c) * kL + (kL - 1 - l1)] = v;
    }
}

// ---------------- dt projection + softplus (K=12: VALU) ----------------
__global__ __launch_bounds__(256)
void ss2d_dt(const float* __restrict__ dbl, const float* __restrict__ dtw,
             const float* __restrict__ dtb, float* __restrict__ dts)
{
    int bk = blockIdx.x;               // 0..7  (b*4+k)
    int c  = blockIdx.y;               // 0..383
    int k  = bk & 3;
    float w[kR];
    #pragma unroll
    for (int r = 0; r < kR; ++r) w[r] = dtw[((long)k * kC + c) * kR + r];
    float bias = dtb[k * kC + c];
    const float* dsrc = dbl + (long)bk * 44 * kL;
    float* dd = dts + ((long)bk * kC + c) * kL;
    for (int l = threadIdx.x; l < kL; l += blockDim.x) {
        float a = bias;
        #pragma unroll
        for (int r = 0; r < kR; ++r) a += w[r] * dsrc[(long)r * kL + l];
        dd[l] = (a > 20.f) ? a : __logf(1.f + __expf(a));
    }
}

// ---------------- chunked parallel scan: pass 1 (chunk summaries) ----------------
__global__ __launch_bounds__(384)
void ss2d_scan_p1(const float* __restrict__ xs, const float* __restrict__ dts,
                  const float* __restrict__ dbl, const float* __restrict__ A_log,
                  float* __restrict__ Pbuf, float* __restrict__ Sbuf)
{
    __shared__ float Bsh[kN][kT];
    int bk = blockIdx.x, g = blockIdx.y, c = threadIdx.x, k = bk & 3;
    const float* dsrc = dbl + (long)bk * 44 * kL + (long)g * kT;
    for (int e = threadIdx.x; e < kN * kT; e += blockDim.x) {
        int n = e / kT, t = e % kT;
        Bsh[n][t] = dsrc[(long)(kR + n) * kL + t];
    }
    __syncthreads();
    float Aa[kN], h[kN], P[kN];
    #pragma unroll
    for (int n = 0; n < kN; ++n) {
        Aa[n] = -__expf(A_log[((long)k * kC + c) * kN + n]);
        h[n] = 0.f; P[n] = 1.f;
    }
    const float* up = xs  + ((long)bk * kC + c) * kL + g * kT;
    const float* dp = dts + ((long)bk * kC + c) * kL + g * kT;
    for (int t = 0; t < kT; ++t) {
        float u = up[t], dt = dp[t];
        float du = dt * u;
        #pragma unroll
        for (int n = 0; n < kN; ++n) {
            float a = __expf(dt * Aa[n]);
            P[n] *= a;
            h[n] = h[n] * a + du * Bsh[n][t];
        }
    }
    long o = (((long)bk * kC + c) * kG + g) * kN;
    #pragma unroll
    for (int n = 0; n < kN; ++n) { Pbuf[o + n] = P[n]; Sbuf[o + n] = h[n]; }
}

// ---------------- pass 2: inter-chunk scan (49152 indep. recurrences of len 24) --
__global__ void ss2d_scan_p2(const float* __restrict__ Pbuf, const float* __restrict__ Sbuf,
                             float* __restrict__ hinit)
{
    long t = (long)blockIdx.x * blockDim.x + threadIdx.x;  // over B*K*C*N
    if (t >= (long)kB * kK * kC * kN) return;
    long base = (t / kN) * ((long)kG * kN) + (t % kN);
    float h = 0.f;
    for (int g = 0; g < kG; ++g) {
        long idx = base + (long)g * kN;
        hinit[idx] = h;
        h = Sbuf[idx] + Pbuf[idx] * h;
    }
}

// ---------------- pass 3: replay with correct init + output y ----------------
__global__ __launch_bounds__(384)
void ss2d_scan_p3(const float* __restrict__ xs, const float* __restrict__ dts,
                  const float* __restrict__ dbl, const float* __restrict__ A_log,
                  const float* __restrict__ Ds, const float* __restrict__ hinit,
                  float* __restrict__ ys)
{
    __shared__ float Bsh[kN][kT];
    __shared__ float Csh[kN][kT];
    int bk = blockIdx.x, g = blockIdx.y, c = threadIdx.x, k = bk & 3;
    const float* dsrc = dbl + (long)bk * 44 * kL + (long)g * kT;
    for (int e = threadIdx.x; e < kN * kT; e += blockDim.x) {
        int n = e / kT, t = e % kT;
        Bsh[n][t] = dsrc[(long)(kR + n) * kL + t];
        Csh[n][t] = dsrc[(long)(kR + kN + n) * kL + t];
    }
    __syncthreads();
    float Aa[kN], h[kN];
    long ho = (((long)bk * kC + c) * kG + g) * kN;
    #pragma unroll
    for (int n = 0; n < kN; ++n) {
        Aa[n] = -__expf(A_log[((long)k * kC + c) * kN + n]);
        h[n] = hinit[ho + n];
    }
    float dsc = Ds[k * kC + c];
    const float* up = xs  + ((long)bk * kC + c) * kL + g * kT;
    const float* dp = dts + ((long)bk * kC + c) * kL + g * kT;
    float* yp = ys + ((long)bk * kC + c) * kL + g * kT;
    for (int t = 0; t < kT; ++t) {
        float u = up[t], dt = dp[t];
        float du = dt * u;
        float y = dsc * u;
        #pragma unroll
        for (int n = 0; n < kN; ++n) {
            float a = __expf(dt * Aa[n]);
            h[n] = h[n] * a + du * Bsh[n][t];
            y += h[n] * Csh[n][t];
        }
        yp[t] = y;
    }
}

// ---------------- merge 4 directions (inverse permutations) ----------------
__global__ void ss2d_merge(const float* __restrict__ ys, const float* __restrict__ mw,
                           float* __restrict__ merged)
{
    int bc = blockIdx.x;               // b*C + c
    int b = bc / kC, c = bc % kC;
    float w0 = mw[0], w1 = mw[1], w2 = mw[2], w3 = mw[3];
    const float* y0 = ys + ((long)(b * kK + 0) * kC + c) * kL;
    const float* y1 = ys + ((long)(b * kK + 1) * kC + c) * kL;
    const float* y2 = ys + ((long)(b * kK + 2) * kC + c) * kL;
    const float* y3 = ys + ((long)(b * kK + 3) * kC + c) * kL;
    for (int l = threadIdx.x; l < kL; l += blockDim.x) {
        int yy = l / kW, xx = l % kW;
        int l1 = xx * kH + yy;
        float v = w0 * y0[l] + w1 * y1[l1] + w2 * y2[kL - 1 - l] + w3 * y3[kL - 1 - l1];
        merged[((long)b * kL + l) * kC + c] = v;
    }
}

// ---------------- LN (output) + SiLU gating ----------------
__global__ __launch_bounds__(384)
void ss2d_ln_out_silu(const float* __restrict__ merged, const float* __restrict__ g,
                      const float* __restrict__ bb, const float* __restrict__ xz,
                      float* __restrict__ yact)
{
    __shared__ float red[384];
    __shared__ float stat;
    int p = blockIdx.x, c = threadIdx.x;
    float v = merged[(long)p * kC + c];
    red[c] = v;
    float s = ss2d_block_reduce(red, c, 384);
    if (c == 0) stat = s / (float)kC;
    __syncthreads();
    float mean = stat;
    float e = v - mean;
    __syncthreads();
    red[c] = e * e;
    s = ss2d_block_reduce(red, c, 384);
    if (c == 0) stat = s / (float)kC;
    __syncthreads();
    float var = stat;
    float ln = e * rsqrtf(var + 1e-5f) * g[c] + bb[c];
    float z = xz[(long)p * (2 * kC) + kC + c];
    float sz = z / (1.f + __expf(-z));
    yact[(long)p * kC + c] = ln * sz;
}

// ---------------- launch ----------------
extern "C" void kernel_launch(void* const* d_in, const int* in_sizes, int n_in,
                              void* d_out, int out_size, void* d_ws, size_t ws_size,
                              hipStream_t stream)
{
    const float* x         = (const float*)d_in[0];
    const float* ln_in_g   = (const float*)d_in[1];
    const float* ln_in_b   = (const float*)d_in[2];
    const float* in_proj_w = (const float*)d_in[3];
    const float* conv_w    = (const float*)d_in[4];
    const float* conv_b    = (const float*)d_in[5];
    const float* x_proj_w  = (const float*)d_in[6];
    const float* dt_proj_w = (const float*)d_in[7];
    const float* dt_proj_b = (const float*)d_in[8];
    const float* A_log     = (const float*)d_in[9];
    const float* Ds        = (const float*)d_in[10];
    const float* merge_w   = (const float*)d_in[11];
    const float* ln_out_g  = (const float*)d_in[12];
    const float* ln_out_b  = (const float*)d_in[13];
    const float* out_proj_w= (const float*)d_in[14];
    float* out = (float*)d_out;

    // workspace carve-up (f32 elements)
    float* w = (float*)d_ws;
    float* h      = w; w += (long)kP * kD;             // 4608*192
    float* xz     = w; w += (long)kP * 2 * kC;         // 4608*768
    float* xconv  = w; w += (long)kB * kC * kL;        // conv output (NCHW)
    float* xs     = w; w += (long)kB * kK * kC * kL;   // 4 scan dirs
    float* dbl    = w; w += (long)kB * kK * 44 * kL;   // x_proj output
    float* dts    = w; w += (long)kB * kK * kC * kL;   // softplus(dt)
    float* Pbuf   = w; w += (long)kB * kK * kC * kG * kN;
    float* Sbuf   = w; w += (long)kB * kK * kC * kG * kN;
    float* hinit  = w; w += (long)kB * kK * kC * kG * kN;
    float* ys     = w; w += (long)kB * kK * kC * kL;
    float* merged = w; w += (long)kB * kL * kC;
    float* yact   = w; w += (long)kB * kL * kC;
    float* xpw    = w; w += (long)kK * TILE_M * kC;    // padded x_proj_w

    // 0) zero-pad x_proj_w so the x_proj GEMM can stage full 128-row tiles
    ss2d_pad_xprojw<<<dim3((TILE_M * kC + 255) / 256, kK), 256, 0, stream>>>(x_proj_w, xpw);

    // 1) input LN
    ss2d_ln_in<<<kP, 192, 0, stream>>>(x, ln_in_g, ln_in_b, h);

    // 2) in_proj: xz[4608,768] = h[4608,192] x W^T   (WMMA, BT, no resid)
    ss2d_gemm_wmma<1, 0><<<dim3(2 * kC / TILE_N, kP / TILE_M, 1), 256, 0, stream>>>(
        h, 0, 1, in_proj_w, 0, xz, 0, nullptr,
        kP, 2 * kC, kD, kD, kD, 2 * kC);

    // 3) depthwise conv + SiLU
    ss2d_conv_silu<<<kP, kC, 0, stream>>>(xz, conv_w, conv_b, xconv);

    // 4) build 4 scan directions
    ss2d_build_xs<<<kB * kC, 256, 0, stream>>>(xconv, xs);

    // 5) x_proj (batched over b*k): dbl[44,2304] = W_k[44,384] x xs[384,2304]
    //    (WMMA, B non-transposed, no resid)
    ss2d_gemm_wmma<0, 0><<<dim3(kL / TILE_N, 1, kB * kK), 256, 0, stream>>>(
        xpw, (long)TILE_M * kC, kK, xs, (long)kC * kL,
        dbl, (long)44 * kL, nullptr,
        44, kL, kC, kC, kL, kL);

    // 6) dt projection + softplus
    ss2d_dt<<<dim3(kB * kK, kC), 256, 0, stream>>>(dbl, dt_proj_w, dt_proj_b, dts);

    // 7-9) chunked parallel scan
    ss2d_scan_p1<<<dim3(kB * kK, kG), kC, 0, stream>>>(xs, dts, dbl, A_log, Pbuf, Sbuf);
    {
        long tot = (long)kB * kK * kC * kN;
        ss2d_scan_p2<<<(int)((tot + 255) / 256), 256, 0, stream>>>(Pbuf, Sbuf, hinit);
    }
    ss2d_scan_p3<<<dim3(kB * kK, kG), kC, 0, stream>>>(xs, dts, dbl, A_log, Ds, hinit, ys);

    // 10) merge directions
    ss2d_merge<<<kB * kC, 256, 0, stream>>>(ys, merge_w, merged);

    // 11) output LN + SiLU gate
    ss2d_ln_out_silu<<<kP, kC, 0, stream>>>(merged, ln_out_g, ln_out_b, xz, yact);

    // 12) out_proj + residual: out[4608,192] = yact[4608,384] x W^T + x
    //     (WMMA, BT, fused residual)
    ss2d_gemm_wmma<1, 1><<<dim3(kD / TILE_N, kP / TILE_M, 1), 256, 0, stream>>>(
        yact, 0, 1, out_proj_w, 0, out, 0, x,
        kP, kD, kC, kC, kC, kD);
}
